// PHA_RNN_simple_66228395704626
// MI455X (gfx1250) — matrix operand order repaired and checked by reading, not compiled, not run on hardware
//
#include <hip/hip_runtime.h>
#include <hip/hip_bf16.h>

// ---------------------------------------------------------------------------
// Model constants
// ---------------------------------------------------------------------------
#define B_   32
#define T_   1000
#define H_   256
#define NF_  513
#define BT_  (B_ * T_)          // 32000
#define K_IN 1026               // 2*NF
#define K_IN_PAD 1056           // padded to multiple of 32
#define N3H  768                // 3*H
#define ND_PAD 528              // 513 padded to multiple of 16
#define LDH_BF 264              // padded bf16 h row (16B-aligned, bank-spread)

typedef __attribute__((ext_vector_type(16))) __bf16 v16bf;
typedef __attribute__((ext_vector_type(8)))  float  v8f;

union FragBF {
  v16bf v;
  uint4 q[2];
};

__device__ __forceinline__ v8f wmma_bf16(v16bf a, v16bf b, v8f c) {
  // D = A(16x32 bf16) * B(32x16 bf16) + C(16x16 f32)
  return __builtin_amdgcn_wmma_f32_16x16x32_bf16(false, a, false, b,
                                                 (short)0, c, false, false);
}

// ---------------------------------------------------------------------------
// Stage input: concat(real, imag) -> bf16 [BT, K_IN_PAD], zero padded
// ---------------------------------------------------------------------------
__global__ __launch_bounds__(256) void prep_input(const float* __restrict__ x,
                                                  __bf16* __restrict__ dst) {
  long long i = (long long)blockIdx.x * 256 + threadIdx.x;
  const long long total = (long long)BT_ * K_IN_PAD;
  if (i >= total) return;
  int row = (int)(i / K_IN_PAD);
  int c   = (int)(i % K_IN_PAD);
  int b = row / T_;
  int t = row % T_;
  float v = 0.f;
  if (c < NF_) {
    v = x[(((size_t)b * 2 + 0) * T_ + t) * NF_ + c];
  } else if (c < 2 * NF_) {
    v = x[(((size_t)b * 2 + 1) * T_ + t) * NF_ + (c - NF_)];
  }
  dst[i] = (__bf16)v;
}

// ---------------------------------------------------------------------------
// Weight convert + zero-pad: f32 [Ns,Ks] -> bf16 [Nd,Kd]
// ---------------------------------------------------------------------------
__global__ __launch_bounds__(256) void convert_pad(const float* __restrict__ src,
                                                   __bf16* __restrict__ dst,
                                                   int Ns, int Ks, int Nd, int Kd) {
  long long i = (long long)blockIdx.x * 256 + threadIdx.x;
  long long total = (long long)Nd * Kd;
  if (i >= total) return;
  int n = (int)(i / Kd);
  int k = (int)(i % Kd);
  float v = (n < Ns && k < Ks) ? src[(size_t)n * Ks + k] : 0.f;
  dst[i] = (__bf16)v;
}

// ---------------------------------------------------------------------------
// Generic WMMA GEMM:  C[M,N] = A[M,K](bf16) * W[N,K]^T(bf16) + bias[N](f32)
// One 16x16 tile per wave; 8 waves/block share one M-tile.
// OUT_MODE 0: f32 out, 1: bf16 out
// ---------------------------------------------------------------------------
template <int OUT_MODE>
__global__ __launch_bounds__(256) void gemm_wmma(const __bf16* __restrict__ A,
                                                 const __bf16* __restrict__ W,
                                                 const float* __restrict__ bias,
                                                 void* __restrict__ Cout,
                                                 int K, int ldA, int ldC,
                                                 int Nreal, int n_tiles) {
  const int lane = threadIdx.x & 31;
  const int wave = threadIdx.x >> 5;
  const int nb = blockIdx.y * 8 + wave;
  if (nb >= n_tiles) return;
  const int m0 = blockIdx.x * 16;
  const int n0 = nb * 16;
  const int lr = lane & 15;
  const int hi = lane >> 4;

  const __bf16* Arow = A + (size_t)(m0 + lr) * ldA + (hi << 4);
  const __bf16* Wrow = W + (size_t)(n0 + lr) * K + (hi << 4);

  v8f acc;
#pragma unroll
  for (int i = 0; i < 8; i++) acc[i] = 0.f;

  for (int k0 = 0; k0 < K; k0 += 32) {
    FragBF a, b;
    a.q[0] = *(const uint4*)(Arow + k0);
    a.q[1] = *(const uint4*)(Arow + k0 + 8);
    b.q[0] = *(const uint4*)(Wrow + k0);
    b.q[1] = *(const uint4*)(Wrow + k0 + 8);
    acc = wmma_bf16(a.v, b.v, acc);
  }

  const int col = n0 + lr;
  if (col >= Nreal) return;
  const float bv = bias[col];
  const int rbase = m0 + (hi << 3);
  if (OUT_MODE == 0) {
    float* C = (float*)Cout;
#pragma unroll
    for (int i = 0; i < 8; i++)
      C[(size_t)(rbase + i) * ldC + col] = acc[i] + bv;
  } else {
    __bf16* C = (__bf16*)Cout;
#pragma unroll
    for (int i = 0; i < 8; i++)
      C[(size_t)(rbase + i) * ldC + col] = (__bf16)(acc[i] + bv);
  }
}

// ---------------------------------------------------------------------------
// Persistent single-workgroup GRU scan (512 threads = 16 waves, 4 waves/SIMD).
//
// Wave s owns hidden units j in [16s, 16s+16) and computes their r/z/n gate
// tiles (n-tiles {s, s+16, s+32}) so gate fusion is all in registers.
// whh K-low half (k<128) of the wave's 3 tiles: resident in 96 VGPRs.
// whh K-high half: staged once into LDS in exact B-fragment layout (192 KB),
// streamed with conflict-free ds_load_b128.
// h: bf16 double buffer in LDS (A-fragment friendly), h_old in registers.
// ---------------------------------------------------------------------------
#define SB_BYTES  (48 * 4 * 32 * 32)               // 196608: [nb][kc][lane][32B]
#define HB_ELEMS  (2 * 32 * LDH_BF)                // bf16 elements
#define SCAN_SMEM_BYTES (SB_BYTES + HB_ELEMS * 2)  // 230400

__global__ __launch_bounds__(512) void gru_scan(const float* __restrict__ xproj,
                                                const __bf16* __restrict__ whh,
                                                const float* __restrict__ bhh,
                                                __bf16* __restrict__ hseq) {
  extern __shared__ unsigned char smem_raw[];
  __bf16* sB = (__bf16*)smem_raw;                  // staged B fragments (K hi half)
  __bf16* hb = (__bf16*)(smem_raw + SB_BYTES);     // [2][32][LDH_BF]

  const int tid  = threadIdx.x;
  const int lane = tid & 31;
  const int s    = tid >> 5;     // wave 0..15 = hidden slice
  const int lr = lane & 15;
  const int hi = lane >> 4;

  // ---- init: zero h (both buffers) ----
  for (int i = tid; i < HB_ELEMS; i += 512) hb[i] = (__bf16)0.f;

  // ---- stage K-high half of whh into LDS in B-fragment layout ----
  // fragment (nb, kc): lane L holds 16 bf16 at K = 128 + kc*32 + 16*(L>>4)
  for (int u = tid; u < 48 * 4 * 32; u += 512) {
    int lane_ = u & 31;
    int slot  = u >> 5;          // nb*4 + kc
    int nb = slot >> 2;
    int kc = slot & 3;
    const __bf16* src = whh + (size_t)(16 * nb + (lane_ & 15)) * H_ +
                        128 + kc * 32 + ((lane_ >> 4) << 4);
    uint4* dst = (uint4*)(sB) + (size_t)u * 2;
    dst[0] = *(const uint4*)(src);
    dst[1] = *(const uint4*)(src + 8);
  }

  // ---- resident K-low half of this wave's 3 gate tiles ----
  FragBF Bres[3][4];
#pragma unroll
  for (int g = 0; g < 3; g++) {
    const int nb = s + 16 * g;
    const __bf16* Wr = whh + (size_t)(16 * nb + lr) * H_ + (hi << 4);
#pragma unroll
    for (int kk = 0; kk < 4; kk++) {
      Bres[g][kk].q[0] = *(const uint4*)(Wr + kk * 32);
      Bres[g][kk].q[1] = *(const uint4*)(Wr + kk * 32 + 8);
    }
  }

  // per-lane recurrent biases (hidden j = 16s + lr)
  const int j = 16 * s + lr;
  const float bhr = bhh[j];
  const float bhz = bhh[H_ + j];
  const float bhn = bhh[2 * H_ + j];

  // h_old in registers: Hreg[m][i] = h[b= m*16 + i + 8*hi][j]
  float Hreg[2][8];
#pragma unroll
  for (int m = 0; m < 2; m++)
#pragma unroll
    for (int i = 0; i < 8; i++) Hreg[m][i] = 0.f;

  __syncthreads();

  int cur = 0;
  for (int t = 0; t < T_; t++) {
    // ---- prefetch xproj for this step (independent of h) ----
    float xpr[2][8], xpz[2][8], xpn[2][8];
#pragma unroll
    for (int m = 0; m < 2; m++)
#pragma unroll
      for (int i = 0; i < 8; i++) {
        const int b = m * 16 + i + 8 * hi;
        const size_t row = (size_t)b * T_ + t;
        xpr[m][i] = xproj[row * N3H + j];
        xpz[m][i] = xproj[row * N3H + H_ + j];
        xpn[m][i] = xproj[row * N3H + 2 * H_ + j];
      }

    // ---- WMMA: hp(gate tiles) = h @ whh^T ----
    const __bf16* hbc = hb + cur * 32 * LDH_BF;
    v8f D[2][3];
#pragma unroll
    for (int m = 0; m < 2; m++)
#pragma unroll
      for (int g = 0; g < 3; g++)
#pragma unroll
        for (int i = 0; i < 8; i++) D[m][g][i] = 0.f;

#pragma unroll
    for (int m = 0; m < 2; m++) {
      const __bf16* Ar = hbc + (m * 16 + lr) * LDH_BF + (hi << 4);
      // K-low half: B in registers
#pragma unroll
      for (int kk = 0; kk < 4; kk++) {
        FragBF a;
        a.q[0] = *(const uint4*)(Ar + kk * 32);
        a.q[1] = *(const uint4*)(Ar + kk * 32 + 8);
        D[m][0] = wmma_bf16(a.v, Bres[0][kk].v, D[m][0]);
        D[m][1] = wmma_bf16(a.v, Bres[1][kk].v, D[m][1]);
        D[m][2] = wmma_bf16(a.v, Bres[2][kk].v, D[m][2]);
      }
      // K-high half: B streamed from LDS (conflict-free layout)
#pragma unroll
      for (int kc = 0; kc < 4; kc++) {
        FragBF a;
        a.q[0] = *(const uint4*)(Ar + 128 + kc * 32);
        a.q[1] = *(const uint4*)(Ar + 128 + kc * 32 + 8);
#pragma unroll
        for (int g = 0; g < 3; g++) {
          const int slot = (s + 16 * g) * 4 + kc;
          const uint4* bp = (const uint4*)(sB) + ((size_t)slot * 32 + lane) * 2;
          FragBF bfr;
          bfr.q[0] = bp[0];
          bfr.q[1] = bp[1];
          D[m][g] = wmma_bf16(a.v, bfr.v, D[m][g]);
        }
      }
    }

    // ---- gate fusion in registers, state update ----
    __bf16* hbn = hb + (1 - cur) * 32 * LDH_BF;
#pragma unroll
    for (int m = 0; m < 2; m++)
#pragma unroll
      for (int i = 0; i < 8; i++) {
        const int b = m * 16 + i + 8 * hi;
        const size_t row = (size_t)b * T_ + t;
        const float hr = D[m][0][i] + bhr;
        const float hz = D[m][1][i] + bhz;
        const float hn = D[m][2][i] + bhn;
        const float r = 1.f / (1.f + __expf(-(xpr[m][i] + hr)));
        const float z = 1.f / (1.f + __expf(-(xpz[m][i] + hz)));
        const float n = tanhf(xpn[m][i] + r * hn);
        const float hnew = (1.f - z) * n + z * Hreg[m][i];
        Hreg[m][i] = hnew;
        hbn[b * LDH_BF + j] = (__bf16)hnew;
        hseq[row * H_ + j] = (__bf16)hnew;
      }
    __syncthreads();
    cur ^= 1;
  }
}

// ---------------------------------------------------------------------------
// Host side
// ---------------------------------------------------------------------------
static inline size_t alignup(size_t x) { return (x + 255) & ~(size_t)255; }

extern "C" void kernel_launch(void* const* d_in, const int* in_sizes, int n_in,
                              void* d_out, int out_size, void* d_ws, size_t ws_size,
                              hipStream_t stream) {
  const float* x      = (const float*)d_in[0];
  const float* ds_w   = (const float*)d_in[1];
  const float* ds_b   = (const float*)d_in[2];
  const float* g1_wih = (const float*)d_in[3];
  const float* g1_whh = (const float*)d_in[4];
  const float* g1_bih = (const float*)d_in[5];
  const float* g1_bhh = (const float*)d_in[6];
  const float* gr_wih = (const float*)d_in[7];
  const float* gr_whh = (const float*)d_in[8];
  const float* gr_bih = (const float*)d_in[9];
  const float* gr_bhh = (const float*)d_in[10];
  const float* gi_wih = (const float*)d_in[11];
  const float* gi_whh = (const float*)d_in[12];
  const float* gi_bih = (const float*)d_in[13];
  const float* gi_bhh = (const float*)d_in[14];
  const float* dr_w   = (const float*)d_in[15];
  const float* dr_b   = (const float*)d_in[16];
  const float* di_w   = (const float*)d_in[17];
  const float* di_b   = (const float*)d_in[18];

  // --- workspace carve ---
  unsigned char* ws = (unsigned char*)d_ws;
  size_t off = 0;
  auto carve = [&](size_t bytes) { unsigned char* p = ws + off; off = alignup(off + bytes); return p; };

  __bf16* in_bf    = (__bf16*)carve((size_t)BT_ * K_IN_PAD * 2);
  __bf16* so_bf    = (__bf16*)carve((size_t)BT_ * H_ * 2);
  float*  xproj    = (float*)carve((size_t)BT_ * N3H * 4);
  __bf16* h1seq    = (__bf16*)carve((size_t)BT_ * H_ * 2);
  __bf16* tseq     = (__bf16*)carve((size_t)BT_ * H_ * 2);
  __bf16* w_ds     = (__bf16*)carve((size_t)H_ * K_IN_PAD * 2);
  __bf16* w_g1ih   = (__bf16*)carve((size_t)N3H * H_ * 2);
  __bf16* w_g1hh   = (__bf16*)carve((size_t)N3H * H_ * 2);
  __bf16* w_grih   = (__bf16*)carve((size_t)N3H * H_ * 2);
  __bf16* w_grhh   = (__bf16*)carve((size_t)N3H * H_ * 2);
  __bf16* w_giih   = (__bf16*)carve((size_t)N3H * H_ * 2);
  __bf16* w_gihh   = (__bf16*)carve((size_t)N3H * H_ * 2);
  __bf16* w_dr     = (__bf16*)carve((size_t)ND_PAD * H_ * 2);
  __bf16* w_di     = (__bf16*)carve((size_t)ND_PAD * H_ * 2);
  (void)ws_size; (void)in_sizes; (void)n_in; (void)out_size;

  float* out_real = (float*)d_out;
  float* out_imag = out_real + (size_t)B_ * T_ * NF_;

  // --- stage input + weights ---
  {
    long long total = (long long)BT_ * K_IN_PAD;
    prep_input<<<dim3((unsigned)((total + 255) / 256)), dim3(256), 0, stream>>>(x, in_bf);
  }
  auto conv = [&](const float* s, __bf16* d, int Ns, int Ks, int Nd, int Kd) {
    long long total = (long long)Nd * Kd;
    convert_pad<<<dim3((unsigned)((total + 255) / 256)), dim3(256), 0, stream>>>(s, d, Ns, Ks, Nd, Kd);
  };
  conv(ds_w,   w_ds,   H_,  K_IN, H_,     K_IN_PAD);
  conv(g1_wih, w_g1ih, N3H, H_,   N3H,    H_);
  conv(g1_whh, w_g1hh, N3H, H_,   N3H,    H_);
  conv(gr_wih, w_grih, N3H, H_,   N3H,    H_);
  conv(gr_whh, w_grhh, N3H, H_,   N3H,    H_);
  conv(gi_wih, w_giih, N3H, H_,   N3H,    H_);
  conv(gi_whh, w_gihh, N3H, H_,   N3H,    H_);
  conv(dr_w,   w_dr,   NF_, H_,   ND_PAD, H_);
  conv(di_w,   w_di,   NF_, H_,   ND_PAD, H_);

  const dim3 blk(256);
  const unsigned mtiles = BT_ / 16;  // 2000

  // shared dense: [BT,1056] -> bf16 [BT,256]
  gemm_wmma<1><<<dim3(mtiles, 2), blk, 0, stream>>>(in_bf, w_ds, ds_b, so_bf,
                                                    K_IN_PAD, K_IN_PAD, H_, H_, H_ / 16);
  // GRU1 input proj: -> f32 [BT,768]
  gemm_wmma<0><<<dim3(mtiles, 6), blk, 0, stream>>>(so_bf, w_g1ih, g1_bih, xproj,
                                                    H_, H_, N3H, N3H, N3H / 16);
  gru_scan<<<dim3(1), dim3(512), SCAN_SMEM_BYTES, stream>>>(xproj, w_g1hh, g1_bhh, h1seq);

  // real branch
  gemm_wmma<0><<<dim3(mtiles, 6), blk, 0, stream>>>(h1seq, w_grih, gr_bih, xproj,
                                                    H_, H_, N3H, N3H, N3H / 16);
  gru_scan<<<dim3(1), dim3(512), SCAN_SMEM_BYTES, stream>>>(xproj, w_grhh, gr_bhh, tseq);
  gemm_wmma<0><<<dim3(mtiles, 5), blk, 0, stream>>>(tseq, w_dr, dr_b, out_real,
                                                    H_, H_, NF_, NF_, ND_PAD / 16);

  // imag branch
  gemm_wmma<0><<<dim3(mtiles, 6), blk, 0, stream>>>(h1seq, w_giih, gi_bih, xproj,
                                                    H_, H_, N3H, N3H, N3H / 16);
  gru_scan<<<dim3(1), dim3(512), SCAN_SMEM_BYTES, stream>>>(xproj, w_gihh, gi_bhh, tseq);
  gemm_wmma<0><<<dim3(mtiles, 5), blk, 0, stream>>>(tseq, w_di, di_b, out_imag,
                                                    H_, H_, NF_, NF_, ND_PAD / 16);
}
